// CausalSelfAttention_89532888252836
// MI455X (gfx1250) — matrix operand order repaired and checked by reading, not compiled, not run on hardware
//
#include <hip/hip_runtime.h>
#include <math.h>

typedef unsigned short u16;
typedef unsigned int   u32;

typedef __attribute__((ext_vector_type(16))) __bf16 v16bf;
typedef __attribute__((ext_vector_type(8)))  float  v8f;

union Frag {
    v16bf bf;
    uint4 u4[2];
    u16   s[16];
};

__device__ __forceinline__ u16 f2bf(float f) {
    u32 u = __float_as_uint(f);
    u32 r = u + 0x7FFFu + ((u >> 16) & 1u);   // round-to-nearest-even
    return (u16)(r >> 16);
}

__device__ __forceinline__ v8f wmma_bf16(v16bf a, v16bf b, v8f c) {
    return __builtin_amdgcn_wmma_f32_16x16x32_bf16(
        false, a, false, b, (short)0, c, false, false);
}

// ---------------------------------------------------------------------------
// Tiled bf16 GEMM with fp32 accumulate, software-pipelined global->LDS staging.
//   C[M,N] = A[M,1024] * B[1024,N] + bias
// Block tile 128x128, K-step 32, 256 threads = 8 waves, wave tile 64x32.
// A is fp32 (converted in-loader) or bf16 (u16). B is fp32 weights.
// EPI==0: scatter bf16 into q[BH,T,64] (pre-scaled by 1/sqrt(HD)),
//         k[BH,T,64], vT[BH,64,T]                                  (N=3072)
// EPI==1: write fp32 to fo[M,N]                                    (N=1024)
// ---------------------------------------------------------------------------
template<int EPI, bool A_IS_F32>
__global__ __launch_bounds__(256) void gemm_kernel(
    const void*  __restrict__ Ag,
    const float* __restrict__ Bg,
    const float* __restrict__ bias,
    u16* __restrict__ qo, u16* __restrict__ ko, u16* __restrict__ vTo,
    float* __restrict__ fo, int N)
{
    constexpr int Kd  = 1024;
    constexpr int LDP = 40;               // padded LDS row (bf16 elems) -> conflict-free
    __shared__ u16 As[128 * LDP];
    __shared__ u16 Bs[128 * LDP];

    const int tid  = threadIdx.x;
    const int m0   = blockIdx.y * 128;
    const int n0   = blockIdx.x * 128;
    const int lane = tid & 31;
    const int wid  = tid >> 5;
    const int nl   = lane & 15;
    const int hi   = lane >> 4;
    const int wm   = (wid >> 2) * 64;     // wave M offset (0,64)
    const int wn   = (wid & 3) * 32;      // wave N offset (0,32,64,96)

    v8f acc[4][2];
    #pragma unroll
    for (int a = 0; a < 4; ++a)
        #pragma unroll
        for (int b = 0; b < 2; ++b)
            #pragma unroll
            for (int j = 0; j < 8; ++j) acc[a][b][j] = 0.0f;

    const int arow  = tid >> 1, ahalf = tid & 1;  // A: 128 rows x 2 half-rows of 16
    const int bk    = tid >> 3, bseg  = tid & 7;  // B: 32 k-rows x 8 segs of 16 n

    // register staging buffers for the pipelined fetch
    float4 aF[4];
    uint4  aB[2];
    float4 bF[4];

    auto fetch = [&](int k0) {
        if constexpr (A_IS_F32) {
            const float4* ap = (const float4*)((const float*)Ag +
                               (size_t)(m0 + arow) * Kd + k0 + ahalf * 16);
            aF[0] = ap[0]; aF[1] = ap[1]; aF[2] = ap[2]; aF[3] = ap[3];
        } else {
            const uint4* ap = (const uint4*)((const u16*)Ag +
                              (size_t)(m0 + arow) * Kd + k0 + ahalf * 16);
            aB[0] = ap[0]; aB[1] = ap[1];
        }
        const float4* bp = (const float4*)(Bg + (size_t)(k0 + bk) * N + n0 + bseg * 16);
        bF[0] = bp[0]; bF[1] = bp[1]; bF[2] = bp[2]; bF[3] = bp[3];
    };

    auto stage = [&]() {
        if constexpr (A_IS_F32) {
            u16* dst = &As[arow * LDP + ahalf * 16];
            #pragma unroll
            for (int i = 0; i < 4; ++i) {
                dst[i * 4 + 0] = f2bf(aF[i].x);
                dst[i * 4 + 1] = f2bf(aF[i].y);
                dst[i * 4 + 2] = f2bf(aF[i].z);
                dst[i * 4 + 3] = f2bf(aF[i].w);
            }
        } else {
            uint4* dst = (uint4*)&As[arow * LDP + ahalf * 16];
            dst[0] = aB[0]; dst[1] = aB[1];
        }
        #pragma unroll
        for (int i = 0; i < 4; ++i) {
            Bs[(bseg * 16 + i * 4 + 0) * LDP + bk] = f2bf(bF[i].x);
            Bs[(bseg * 16 + i * 4 + 1) * LDP + bk] = f2bf(bF[i].y);
            Bs[(bseg * 16 + i * 4 + 2) * LDP + bk] = f2bf(bF[i].z);
            Bs[(bseg * 16 + i * 4 + 3) * LDP + bk] = f2bf(bF[i].w);
        }
    };

    fetch(0);
    for (int k0 = 0; k0 < Kd; k0 += 32) {
        stage();
        __syncthreads();
        if (k0 + 32 < Kd) fetch(k0 + 32);   // next slice flies during compute

        Frag fa[4], fb[2];
        #pragma unroll
        for (int mt = 0; mt < 4; ++mt) {
            const u16* p = &As[(wm + mt * 16 + nl) * LDP + hi * 8];
            fa[mt].u4[0] = *(const uint4*)p;
            fa[mt].u4[1] = *(const uint4*)(p + 16);
        }
        #pragma unroll
        for (int nt = 0; nt < 2; ++nt) {
            const u16* p = &Bs[(wn + nt * 16 + nl) * LDP + hi * 8];
            fb[nt].u4[0] = *(const uint4*)p;
            fb[nt].u4[1] = *(const uint4*)(p + 16);
        }
        #pragma unroll
        for (int mt = 0; mt < 4; ++mt)
            #pragma unroll
            for (int nt = 0; nt < 2; ++nt)
                acc[mt][nt] = wmma_bf16(fa[mt].bf, fb[nt].bf, acc[mt][nt]);
        __syncthreads();
    }

    // ---- epilogue ----
    #pragma unroll
    for (int nt = 0; nt < 2; ++nt) {
        const int n  = n0 + wn + nt * 16 + nl;
        const float bv = bias[n];
        if (EPI == 0) {
            const int part = n >> 10;          // 0=q 1=k 2=v (uniform per block)
            const int rem  = n & 1023;
            const int h    = rem >> 6;
            const int d    = rem & 63;
            const float ps = (part == 0) ? 0.125f : 1.0f;  // fold 1/sqrt(64) into q
            #pragma unroll
            for (int mt = 0; mt < 4; ++mt)
                #pragma unroll
                for (int j = 0; j < 8; ++j) {
                    const int m = m0 + wm + mt * 16 + hi * 8 + j;
                    const int b = m >> 11;     // T = 2048
                    const int t = m & 2047;
                    const u16 bf = f2bf((acc[mt][nt][j] + bv) * ps);
                    const size_t hoff = (size_t)(b * 16 + h) * (2048 * 64);
                    if (part == 0)      qo[hoff + (size_t)t * 64 + d]    = bf;
                    else if (part == 1) ko[hoff + (size_t)t * 64 + d]    = bf;
                    else                vTo[hoff + (size_t)d * 2048 + t] = bf;
                }
        } else {
            #pragma unroll
            for (int mt = 0; mt < 4; ++mt)
                #pragma unroll
                for (int j = 0; j < 8; ++j) {
                    const int m = m0 + wm + mt * 16 + hi * 8 + j;
                    fo[(size_t)m * N + n] = acc[mt][nt][j] + bv;
                }
        }
    }
}

// ---------------------------------------------------------------------------
// One 32-key flash-attention chunk. MASK=false for chunks entirely below the
// causal diagonal (no compares/selects emitted at all).
// ---------------------------------------------------------------------------
template<bool MASK>
__device__ __forceinline__ void attn_chunk(
    const u16* __restrict__ Kh, const u16* __restrict__ Vh,
    int k0, int nl, int hi, int q,
    const Frag (&bq)[2], float& m_run, float& l_run, v8f (&o)[4])
{
    // ---- issue ALL global loads for this chunk up front (one clause) ----
    Frag ak[4];                          // [tile*2 + chunk]
    #pragma unroll
    for (int t2 = 0; t2 < 2; ++t2)
        #pragma unroll
        for (int c = 0; c < 2; ++c) {
            const u16* p = Kh + (size_t)(k0 + t2 * 16 + nl) * 64 + 32 * c + hi * 8;
            ak[t2 * 2 + c].u4[0] = *(const uint4*)p;
            ak[t2 * 2 + c].u4[1] = *(const uint4*)(p + 16);
        }
    Frag bv[4];                          // V fragments land during S-wmma+softmax
    #pragma unroll
    for (int dt = 0; dt < 4; ++dt) {
        const u16* vp = Vh + (size_t)(dt * 16 + nl) * 2048 + k0 + hi * 8;
        bv[dt].u4[0] = *(const uint4*)vp;
        bv[dt].u4[1] = *(const uint4*)(vp + 16);
    }

    // ---- S^T = K * Q^T for key tiles [k0,k0+16) and [k0+16,k0+32) ----
    v8f s0, s1;
    #pragma unroll
    for (int j = 0; j < 8; ++j) { s0[j] = 0.0f; s1[j] = 0.0f; }
    s0 = wmma_bf16(ak[0].bf, bq[0].bf, s0);
    s0 = wmma_bf16(ak[1].bf, bq[1].bf, s0);
    s1 = wmma_bf16(ak[2].bf, bq[0].bf, s1);
    s1 = wmma_bf16(ak[3].bf, bq[1].bf, s1);

    // causal mask (boundary chunks only); q pre-scaled so S needs no scaling
    float p0v[8], p1v[8];
    float mloc = -INFINITY;
    #pragma unroll
    for (int j = 0; j < 8; ++j) {
        float v0 = s0[j];
        float v1 = s1[j];
        if (MASK) {
            const int key0 = k0 + hi * 8 + j;   // S^T element: key row, q col
            const int key1 = key0 + 16;
            v0 = (key0 > q) ? -INFINITY : v0;
            v1 = (key1 > q) ? -INFINITY : v1;
        }
        p0v[j] = v0; p1v[j] = v1;
        mloc = fmaxf(mloc, fmaxf(v0, v1));
    }
    mloc = fmaxf(mloc, __shfl_xor(mloc, 16));
    const float m_new = fmaxf(m_run, mloc);
    const float corr  = __expf(m_run - m_new);

    // exponentiate -> bf16 P A-fragment (lane-local by construction)
    Frag pf;
    float lsum = 0.0f;
    #pragma unroll
    for (int j = 0; j < 8; ++j) {
        const float e0 = __expf(p0v[j] - m_new);
        const float e1 = __expf(p1v[j] - m_new);
        lsum += e0 + e1;
        pf.s[j]     = f2bf(e0);
        pf.s[8 + j] = f2bf(e1);
    }
    lsum += __shfl_xor(lsum, 16);
    l_run = l_run * corr + lsum;
    m_run = m_new;

    // rescale O rows (row q_local = hi*8+j; stats live at lane q_local)
    float cf[8];
    #pragma unroll
    for (int j = 0; j < 8; ++j) cf[j] = __shfl(corr, (hi << 3) + j);
    #pragma unroll
    for (int t = 0; t < 4; ++t)
        #pragma unroll
        for (int j = 0; j < 8; ++j) o[t][j] *= cf[j];

    // ---- O += P * V (V fragments already resident) ----
    #pragma unroll
    for (int dt = 0; dt < 4; ++dt)
        o[dt] = wmma_bf16(pf.bf, bv[dt].bf, o[dt]);
}

// ---------------------------------------------------------------------------
// Flash attention: one wave handles one (b,h, 16-query tile).
// Computes S^T = K*Q^T so that two consecutive 16-key S^T accumulators map
// lane-locally onto the A-fragment of P for O += P*V (no cross-lane traffic).
// q layout [BH,T,64] bf16 pre-scaled by 1/8 ; k [BH,T,64] ; v^T [BH,64,T].
// Output y bf16 [B*T, 1024].
// ---------------------------------------------------------------------------
__global__ __launch_bounds__(256) void attn_kernel(
    const u16* __restrict__ Q, const u16* __restrict__ K,
    const u16* __restrict__ VT, u16* __restrict__ Y)
{
    const int lane = threadIdx.x & 31;
    const int wid  = threadIdx.x >> 5;
    const int job  = blockIdx.x * 8 + wid;   // 8192 jobs
    const int qt   = job & 127;              // q tile within head (T/16 = 128)
    const int bh   = job >> 7;               // 0..63
    const int q0   = qt * 16;
    const int nl   = lane & 15;
    const int hi   = lane >> 4;

    const size_t hoff = (size_t)bh * (2048 * 64);
    const u16* Qh = Q  + hoff;
    const u16* Kh = K  + hoff;
    const u16* Vh = VT + hoff;

    // Q^T B-fragments (reduction dim = head dim, 2 chunks of 32)
    Frag bq[2];
    #pragma unroll
    for (int c = 0; c < 2; ++c) {
        const u16* p = Qh + (size_t)(q0 + nl) * 64 + 32 * c + hi * 8;
        bq[c].u4[0] = *(const uint4*)p;
        bq[c].u4[1] = *(const uint4*)(p + 16);
    }

    float m_run = -INFINITY;   // stats for query q0+nl (duplicated in both halves)
    float l_run = 0.0f;
    v8f o[4];
    #pragma unroll
    for (int t = 0; t < 4; ++t)
        #pragma unroll
        for (int j = 0; j < 8; ++j) o[t][j] = 0.0f;

    const int q       = q0 + nl;
    const int nchunks = (q0 + 16 + 31) >> 5; // 32 keys per chunk
    const int nfull   = q0 >> 5;             // chunks entirely below the diagonal

    for (int kc = 0; kc < nfull; ++kc) {
        const int k0  = kc * 32;
        // uniform, clamped, speculative prefetch of next chunk (no exec mask)
        const int kpf = (k0 + 32 < 2048) ? (k0 + 32) : k0;
        __builtin_prefetch(Kh + (size_t)(kpf + lane) * 64, 0, 3);
        __builtin_prefetch(Vh + (size_t)lane * 2048 + kpf, 0, 3);
        attn_chunk<false>(Kh, Vh, k0, nl, hi, q, bq, m_run, l_run, o);
    }
    for (int kc = nfull; kc < nchunks; ++kc)
        attn_chunk<true>(Kh, Vh, kc * 32, nl, hi, q, bq, m_run, l_run, o);

    // normalize + store to y[B*T, 1024] bf16
    float linv[8];
    #pragma unroll
    for (int j = 0; j < 8; ++j) linv[j] = 1.0f / __shfl(l_run, (hi << 3) + j);

    const int b = bh >> 4, h = bh & 15;
    #pragma unroll
    for (int dt = 0; dt < 4; ++dt)
        #pragma unroll
        for (int j = 0; j < 8; ++j) {
            const int t = q0 + hi * 8 + j;
            const int c = h * 64 + dt * 16 + nl;
            Y[(size_t)(b * 2048 + t) * 1024 + c] = f2bf(o[dt][j] * linv[j]);
        }
}

// ---------------------------------------------------------------------------
extern "C" void kernel_launch(void* const* d_in, const int* in_sizes, int n_in,
                              void* d_out, int out_size, void* d_ws, size_t ws_size,
                              hipStream_t stream)
{
    const float* x  = (const float*)d_in[0];   // [4,2048,1024]
    const float* Wa = (const float*)d_in[1];   // [1024,3072]
    const float* ba = (const float*)d_in[2];   // [3072]
    const float* Wp = (const float*)d_in[3];   // [1024,1024]
    const float* bp = (const float*)d_in[4];   // [1024]

    const size_t HE = (size_t)64 * 2048 * 64;  // 8,388,608 elems per tensor
    u16* qb  = (u16*)d_ws;
    u16* kb  = qb  + HE;
    u16* vTb = kb  + HE;
    u16* yb  = vTb + HE;                       // total 64 MB of bf16 scratch

    // 1) QKV projection: [8192,1024] x [1024,3072] + bias -> q*0.125, k, v^T (bf16)
    gemm_kernel<0, true><<<dim3(24, 64), 256, 0, stream>>>(
        (const void*)x, Wa, ba, qb, kb, vTb, nullptr, 3072);

    // 2) Causal flash attention (bf16 WMMA, fp32 online softmax) -> y bf16
    attn_kernel<<<1024, 256, 0, stream>>>(qb, kb, vTb, yb);

    // 3) Output projection: [8192,1024] x [1024,1024] + bias -> fp32 out
    gemm_kernel<1, false><<<dim3(8, 64), 256, 0, stream>>>(
        (const void*)yb, Wp, bp, nullptr, nullptr, nullptr, (float*)d_out, 1024);
}
